// SPYOLOv6Head_71536975282581
// MI455X (gfx1250) — matrix operand order repaired and checked by reading, not compiled, not run on hardware
//
#include <hip/hip_runtime.h>
#include <hip/hip_bf16.h>

typedef __attribute__((ext_vector_type(16))) __bf16 v16bf;
typedef __attribute__((ext_vector_type(8)))  float  v8f;

struct U4 { unsigned int a, b, c, d; };

union Frag {
  v16bf v;
  unsigned int dw[8];
  U4 q[2];
};

// ---- problem sizes ----
#define BSZ   8
#define CIN   256
#define CH    256
#define NYY   80
#define NXX   80
#define NPTS  16384
#define PADW  82
#define NCLS  80

// ---- workspace layout (bf16 elements) ----
#define OFF_STEMW   0u           // [256][256]
#define OFF_CONVW   65536u       // [2][9][256][256]  (s=0 cls, s=1 reg), contiguous in c
#define OFF_PREDC   1245184u     // [80][256]
#define OFF_PREDRO  1265664u     // [16][256] rows 0-3 reg, row 4 obj, rows 5-15 zero
#define OFF_PSTEM   1269760u     // [8][82][82][256] padded NHWC stem
#define PSTEM_ELEMS 13768704u    // 8*82*82*256 - 2048

__device__ __forceinline__ float silu_f(float v) {
  return v / (1.0f + __expf(-v));
}

// A-fragment (16-bit 16x32): base must already point at row + c0 + half*16
__device__ __forceinline__ void load_afrag(Frag& f, const __bf16* base) {
#pragma unroll
  for (int g = 0; g < 8; ++g) {
    const int kb = 2 * g;
    f.dw[g] = *(const unsigned int*)(base + kb);
  }
}

// B-fragment (16-bit 32x16): p points at [K=c0 + half*16] of column N, 32B aligned
__device__ __forceinline__ void load_bfrag(Frag& f, const __bf16* p) {
  f.q[0] = *(const U4*)(p);
  f.q[1] = *(const U4*)(p + 8);
}

__device__ __forceinline__ v8f wmma_bf16(const Frag& a, const Frag& b, v8f c) {
  return __builtin_amdgcn_wmma_f32_16x16x32_bf16(false, a.v, false, b.v,
                                                 (short)0, c, false, false);
}

// 32B-granular XOR swizzle inside a 512B patch row (conflict spreading).
__device__ __forceinline__ unsigned tap_swz(int patch, unsigned byteInRow) {
  unsigned unit = (byteInRow >> 5) & 15u;
  unsigned rem  = byteInRow & 31u;
  unit ^= (unsigned)(patch & 15);
  return (unsigned)patch * 512u + (unit << 5) + rem;
}

// ---------------------------------------------------------------------------
// Kernel 0a: zero padded stem buffer (border must be zero; ws is poisoned)
// ---------------------------------------------------------------------------
__global__ void zero_pstem_kernel(__bf16* __restrict__ ws) {
  unsigned int* p = (unsigned int*)(ws + OFF_PSTEM);
  const unsigned int n = PSTEM_ELEMS / 2u;
  for (unsigned int i = blockIdx.x * blockDim.x + threadIdx.x; i < n;
       i += gridDim.x * blockDim.x)
    p[i] = 0u;
}

// ---------------------------------------------------------------------------
// Kernel 0b: convert / rearrange all weights to bf16 in ws
// ---------------------------------------------------------------------------
__global__ void prep_weights_kernel(const float* __restrict__ stem_w,
                                    const float* __restrict__ cls_conv_w,
                                    const float* __restrict__ reg_conv_w,
                                    const float* __restrict__ cls_pred_w,
                                    const float* __restrict__ reg_pred_w,
                                    const float* __restrict__ obj_pred_w,
                                    __bf16* __restrict__ ws) {
  const unsigned int N_STEMW = 65536u;
  const unsigned int N_CONV  = 1179648u;  // 2*9*256*256
  const unsigned int N_PREDC = 20480u;    // 80*256
  const unsigned int N_PREDR = 4096u;     // 16*256
  const unsigned int TOT = N_STEMW + N_CONV + N_PREDC + N_PREDR;
  for (unsigned int i = blockIdx.x * blockDim.x + threadIdx.x; i < TOT;
       i += gridDim.x * blockDim.x) {
    if (i < N_STEMW) {
      ws[OFF_STEMW + i] = (__bf16)stem_w[i];
    } else if (i < N_STEMW + N_CONV) {
      unsigned int d = i - N_STEMW;
      unsigned int s  = d / 589824u;
      unsigned int r  = d % 589824u;
      unsigned int j  = r / 65536u;
      unsigned int rr = r % 65536u;
      unsigned int o  = rr / 256u;
      unsigned int c  = rr % 256u;
      const float* w = s ? reg_conv_w : cls_conv_w;
      ws[OFF_CONVW + d] = (__bf16)w[o * 2304u + c * 9u + j];
    } else if (i < N_STEMW + N_CONV + N_PREDC) {
      unsigned int d = i - (N_STEMW + N_CONV);
      ws[OFF_PREDC + d] = (__bf16)cls_pred_w[d];
    } else {
      unsigned int d = i - (N_STEMW + N_CONV + N_PREDC);
      unsigned int row = d / 256u, c = d % 256u;
      float v = (row < 4u) ? reg_pred_w[row * 256u + c]
                           : ((row == 4u) ? obj_pred_w[c] : 0.0f);
      ws[OFF_PREDRO + d] = (__bf16)v;
    }
  }
}

// ---------------------------------------------------------------------------
// Kernel 1: stem 1x1 conv GEMM  (256 out-ch x 64 pixels per block, K=256)
//   out = silu(stem_w @ x + b) stored padded NHWC bf16
//   Double-buffered x staging: stage tile k+1 while WMMAs run on tile k.
// ---------------------------------------------------------------------------
__global__ __launch_bounds__(256) void stem_kernel(const float* __restrict__ x,
                                                   const float* __restrict__ stem_b,
                                                   __bf16* __restrict__ ws) {
  __shared__ __bf16 xb[2][64 * 32];  // [buf][pixel][c_local] bf16, 2 x 4 KB

  const __bf16* wsm = ws + OFF_STEMW;
  __bf16* pstem     = ws + OFF_PSTEM;

  const int tile = blockIdx.x;  // 0..799
  const int b    = tile / 100;
  const int hw0  = (tile % 100) * 64;
  const int tid  = threadIdx.x;
  const int wave = tid >> 5;
  const int lane = tid & 31;
  const int half = lane >> 4;
  const int lm   = lane & 15;

  const int rrow = tid >> 3;  // 0..31 channel
  const int rchk = tid & 7;   // 0..7 pixel octet

  auto stage = [&](int ks) {  // fp32 -> bf16 stage of x tile (32 c x 64 px)
    const float* src =
        x + ((size_t)b * CIN + (ks * 32 + rrow)) * (NYY * NXX) + hw0 + rchk * 8;
    __bf16* dst = &xb[ks & 1][0];
#pragma unroll
    for (int e = 0; e < 8; ++e)
      dst[(rchk * 8 + e) * 32 + rrow] = (__bf16)src[e];
  };

  v8f acc[2][4];
#pragma unroll
  for (int o = 0; o < 2; ++o)
#pragma unroll
    for (int t = 0; t < 4; ++t) acc[o][t] = {};

  stage(0);
  __syncthreads();

#pragma unroll 1
  for (int ks = 0; ks < 8; ++ks) {
    if (ks < 7) stage(ks + 1);  // other buffer; overlaps with compute below

    Frag bfr[4];
#pragma unroll
    for (int nt = 0; nt < 4; ++nt)
      load_bfrag(bfr[nt], &xb[ks & 1][(nt * 16 + lm) * 32 + half * 16]);

    Frag afr[2];
#pragma unroll
    for (int ot = 0; ot < 2; ++ot) {
      const int orow = (wave * 2 + ot) * 16 + lm;
      load_afrag(afr[ot], wsm + (size_t)orow * 256 + ks * 32 + half * 16);
    }

#pragma unroll
    for (int nt = 0; nt < 4; ++nt)
#pragma unroll
      for (int ot = 0; ot < 2; ++ot)
        acc[ot][nt] = wmma_bf16(afr[ot], bfr[nt], acc[ot][nt]);

    __syncthreads();  // staging writes visible AND buffer reads retired
  }

  // epilogue: bias + SiLU -> padded NHWC bf16, packed 8 channels -> b128
#pragma unroll
  for (int ot = 0; ot < 2; ++ot) {
    const int obase = (wave * 2 + ot) * 16;
#pragma unroll
    for (int nt = 0; nt < 4; ++nt) {
      const int hw = hw0 + nt * 16 + lm;
      const int yy = hw / NXX, xx = hw % NXX;
      __bf16* dst =
          pstem + (((size_t)b * PADW + (yy + 1)) * PADW + (xx + 1)) * 256;
      U4 pk;
      __bf16* pb = (__bf16*)&pk;
#pragma unroll
      for (int j = 0; j < 8; ++j) {
        const int o = obase + j + half * 8;
        pb[j] = (__bf16)silu_f(acc[ot][nt][j] + stem_b[o]);
      }
      *(U4*)(dst + obase + half * 8) = pk;  // 16B aligned
    }
  }
}

// ---------------------------------------------------------------------------
// Kernel 2: gather + cls/reg conv GEMMs (K=2304) + pred GEMMs + output
//   512 threads (16 waves), 64 patches per block, grid = 256.
//   Per 3x3 tap: 64 patch rows (32 KB bf16) DMA'd into LDS once via
//   global_load_async_to_lds_b128 (double-buffered, overlapped with WMMA).
// ---------------------------------------------------------------------------
__global__ __launch_bounds__(512) void head_kernel(
    const int* __restrict__ idx, const float* __restrict__ cls_conv_b,
    const float* __restrict__ reg_conv_b, const float* __restrict__ cls_pred_b,
    const float* __restrict__ reg_pred_b, const float* __restrict__ obj_pred_b,
    const __bf16* __restrict__ ws, float* __restrict__ out) {
  extern __shared__ char smem[];  // 64 KB: phase A = 2 x 32 KB tap buffers,
                                  // phase B = feat[64][512] bf16 (overlaid)

  const __bf16* convw  = ws + OFF_CONVW;
  const __bf16* predc  = ws + OFF_PREDC;
  const __bf16* predro = ws + OFF_PREDRO;
  const __bf16* pstem  = ws + OFF_PSTEM;

  const int tid  = threadIdx.x;
  const int wave = tid >> 5;  // 0..15
  const int lane = tid & 31;
  const int half = lane >> 4;
  const int lm   = lane & 15;
  const int n0   = blockIdx.x * 64;

  // wave-uniform base row offsets (elements) of the 4 patches this wave DMAs
  unsigned rowofs[4];
#pragma unroll
  for (int q = 0; q < 4; ++q) {
    const int n = n0 + wave * 4 + q;
    const int bb = idx[n * 3 + 0];
    const int yy = idx[n * 3 + 1];
    const int xx = idx[n * 3 + 2];
    rowofs[q] = (((unsigned)bb * PADW + (unsigned)yy) * PADW + (unsigned)xx) * 256u;
  }

  const unsigned ldsbase = (unsigned)(size_t)&smem[0];
  const unsigned long long gbase = (unsigned long long)(size_t)pstem;

  // issue async DMA of tap j's 64 patch rows into buffer (j&1)
  auto issue_tap = [&](int j) {
    const unsigned jofs = ((unsigned)(j / 3) * PADW + (unsigned)(j % 3)) * 256u;
    const unsigned buf  = (unsigned)(j & 1) * 32768u;
#pragma unroll
    for (int q = 0; q < 4; ++q) {
      const int p = wave * 4 + q;
      const unsigned long long g =
          gbase + (unsigned long long)(rowofs[q] + jofs) * 2ull +
          (unsigned long long)((unsigned)lane * 16u);
      const unsigned l = ldsbase + buf + tap_swz(p, (unsigned)lane * 16u);
      asm volatile("global_load_async_to_lds_b128 %0, %1, off"
                   :
                   : "v"(l), "v"(g)
                   : "memory");
    }
  };

  v8f acc[2][4];
#pragma unroll
  for (int o = 0; o < 2; ++o)
#pragma unroll
    for (int t = 0; t < 4; ++t) acc[o][t] = {};

  issue_tap(0);

  // ---- phase A: conv GEMMs, K = 9 taps x 256 channels ----
#pragma unroll 1
  for (int j = 0; j < 9; ++j) {
    if (j < 8) {
      issue_tap(j + 1);  // prefetch next tap (other buffer) during compute
      asm volatile("s_wait_asynccnt 0x4" ::: "memory");  // tap j done
    } else {
      asm volatile("s_wait_asynccnt 0x0" ::: "memory");
    }
    __syncthreads();  // tap j visible to all waves

    const char* tb = smem + (j & 1) * 32768;
#pragma unroll 2
    for (int ci = 0; ci < 8; ++ci) {
      const int c0 = ci * 32;
      Frag bfr[4];
#pragma unroll
      for (int t = 0; t < 4; ++t) {
        const int patch = t * 16 + lm;
        load_bfrag(bfr[t], (const __bf16*)(tb + tap_swz(patch,
                                           (unsigned)(ci * 64 + half * 32))));
      }
      Frag afr[2];
#pragma unroll
      for (int oo = 0; oo < 2; ++oo) {
        const int otg  = wave * 2 + oo;  // 0..31 (cls rows then reg rows)
        const int s    = otg >> 4;
        const int orow = (otg & 15) * 16 + lm;
        load_afrag(afr[oo], convw + (size_t)s * 589824 + (size_t)j * 65536 +
                                (size_t)orow * 256 + c0 + half * 16);
      }
#pragma unroll
      for (int t = 0; t < 4; ++t)
#pragma unroll
        for (int oo = 0; oo < 2; ++oo)
          acc[oo][t] = wmma_bf16(afr[oo], bfr[t], acc[oo][t]);
    }
    __syncthreads();  // all waves done reading buffer (j&1) before reuse
  }

  // ---- bias + SiLU -> feat LDS (bf16), overlays tap buffers ----
  __bf16* feat = (__bf16*)smem;  // [64 patch][512]
#pragma unroll
  for (int oo = 0; oo < 2; ++oo) {
    const int otg   = wave * 2 + oo;
    const int s     = otg >> 4;
    const int obase = (otg & 15) * 16;
    const float* bias = s ? reg_conv_b : cls_conv_b;
#pragma unroll
    for (int t = 0; t < 4; ++t) {
      const int patch = t * 16 + lm;
      U4 pk;
      __bf16* pb = (__bf16*)&pk;
#pragma unroll
      for (int j = 0; j < 8; ++j) {
        const int o = obase + j + half * 8;
        pb[j] = (__bf16)silu_f(acc[oo][t][j] + bias[o]);
      }
      *(U4*)(feat + patch * 512 + s * 256 + obase + half * 8) = pk;
    }
  }
  __syncthreads();

  // ---- phase B: pred GEMMs; 24 (q,t) pairs over 16 waves ----
#pragma unroll 1
  for (int p = wave; p < 24; p += 16) {
    const int q = p >> 2;  // 0..4 cls o-tiles, 5 = reg/obj tile
    const int t = p & 3;
    const int sfeat = (q < 5) ? 0 : 1;
    const __bf16* arow = (q < 5) ? (predc + (size_t)(q * 16 + lm) * 256)
                                 : (predro + (size_t)lm * 256);
    v8f d = {};
#pragma unroll
    for (int ci = 0; ci < 8; ++ci) {
      const int c0 = ci * 32;
      Frag afr, bfr;
      load_afrag(afr, arow + c0 + half * 16);
      load_bfrag(bfr,
                 feat + ((t * 16 + lm) * 512 + sfeat * 256 + c0 + half * 16));
      d = wmma_bf16(afr, bfr, d);
    }
    const int pg = n0 + t * 16 + lm;
    float* orow = out + (size_t)pg * 85;
#pragma unroll
    for (int j = 0; j < 8; ++j) {
      const int M = j + half * 8;
      if (q < 5) {
        const int o = q * 16 + M;  // always < 80
        orow[5 + o] = d[j] + cls_pred_b[o];
      } else {
        if (M < 4)       orow[M] = d[j] + reg_pred_b[M];
        else if (M == 4) orow[4] = d[j] + obj_pred_b[0];
      }
    }
  }
}

// ---------------------------------------------------------------------------
extern "C" void kernel_launch(void* const* d_in, const int* in_sizes, int n_in,
                              void* d_out, int out_size, void* d_ws,
                              size_t ws_size, hipStream_t stream) {
  const float* x          = (const float*)d_in[0];
  const int*   indices    = (const int*)d_in[1];
  const float* stem_w     = (const float*)d_in[2];
  const float* stem_b     = (const float*)d_in[3];
  const float* cls_conv_w = (const float*)d_in[4];
  const float* cls_conv_b = (const float*)d_in[5];
  const float* reg_conv_w = (const float*)d_in[6];
  const float* reg_conv_b = (const float*)d_in[7];
  const float* cls_pred_w = (const float*)d_in[8];
  const float* cls_pred_b = (const float*)d_in[9];
  const float* reg_pred_w = (const float*)d_in[10];
  const float* reg_pred_b = (const float*)d_in[11];
  const float* obj_pred_w = (const float*)d_in[12];
  const float* obj_pred_b = (const float*)d_in[13];
  float* out = (float*)d_out;
  __bf16* ws = (__bf16*)d_ws;

  zero_pstem_kernel<<<2048, 256, 0, stream>>>(ws);
  prep_weights_kernel<<<2048, 256, 0, stream>>>(stem_w, cls_conv_w, reg_conv_w,
                                                cls_pred_w, reg_pred_w,
                                                obj_pred_w, ws);
  stem_kernel<<<800, 256, 0, stream>>>(x, stem_b, ws);
  head_kernel<<<256, 512, 65536, stream>>>(indices, cls_conv_b, reg_conv_b,
                                           cls_pred_b, reg_pred_b, obj_pred_b,
                                           ws, out);
}
